// ENVM_89258010345626
// MI455X (gfx1250) — matrix operand order repaired and checked by reading, not compiled, not run on hardware
//
#include <hip/hip_runtime.h>

// ---------------------------------------------------------------------------
// MI455X (gfx1250): WMMA f16 GEMMs (A + B staged in registers) + f32 atomic GCN
// ---------------------------------------------------------------------------

typedef __attribute__((ext_vector_type(16))) _Float16 v16h;
typedef __attribute__((ext_vector_type(8)))  float    v8f;

union AFrag { v16h v; uint4 q[2]; };

#define NNODES 50000   // multiple of 16: GEMM row tiling is exact
#define NEDGES 800000
#define DH     128

// ---------------------------------------------------------------------------
// Weight transpose + f32->f16 convert: Wt[n][k] = W[k][n], zero-padded to padN
// ---------------------------------------------------------------------------
__global__ void ENVM_wt_f16(const float* __restrict__ W, _Float16* __restrict__ Wt,
                            int rows /*K*/, int cols /*N*/, int padN) {
    int t = blockIdx.x * blockDim.x + threadIdx.x;
    int total = padN * rows;
    if (t >= total) return;
    int n = t / rows;
    int k = t - n * rows;
    float v = (n < cols) ? W[(size_t)k * cols + n] : 0.0f;
    Wt[(size_t)n * rows + k] = (_Float16)v;
}

// ---------------------------------------------------------------------------
// Degree / normalization
// ---------------------------------------------------------------------------
__global__ void ENVM_deg_init(float* __restrict__ deg, int n) {
    int t = blockIdx.x * blockDim.x + threadIdx.x;
    if (t < n) deg[t] = 1.0f;            // self loop
}
__global__ void ENVM_deg_edges(const int* __restrict__ dst, float* __restrict__ deg, int e) {
    int t = blockIdx.x * blockDim.x + threadIdx.x;
    if (t < e) atomicAdd(&deg[dst[t]], 1.0f);
}
__global__ void ENVM_dis(const float* __restrict__ deg, float* __restrict__ dis, int n) {
    int t = blockIdx.x * blockDim.x + threadIdx.x;
    if (t < n) dis[t] = rsqrtf(deg[t]);
}
// wnorm[e] = dis[src]*dis[dst]  (hoists two gathers out of the hot agg loop)
__global__ void ENVM_enorm(const int* __restrict__ src, const int* __restrict__ dst,
                           const float* __restrict__ dis, float* __restrict__ wnorm, int e) {
    int t = blockIdx.x * blockDim.x + threadIdx.x;
    if (t < e) wnorm[t] = dis[src[t]] * dis[dst[t]];
}

// ---------------------------------------------------------------------------
// WMMA GEMM: C[M x Nout] = A[M x K] * Bt^T  (Bt stored [Nout][K], f16)
//   A16 = A is f16 (else f32 converted on load), KF = K/32.
//   A fragments loaded ONCE into registers; per n-tile, ALL B fragments are
//   preloaded into registers (one deep load clause, staggered waits) before
//   the chained v_wmma_f32_16x16x32_f16 accumulation.
//   Epilogue variants are compile-time; M must be a multiple of 16 so the
//   store loop is guard-free.
// One block = 16 rows; wave w handles n-tiles w, w+8, ...
// ---------------------------------------------------------------------------
template<bool A16, int KF, bool BIAS, bool R2, bool RELU, bool OUT16>
__global__ __launch_bounds__(256)
void ENVM_gemm_wmma(const void* __restrict__ Aptr, int lda,
                    const _Float16* __restrict__ Bt, int Nout,
                    float* __restrict__ Cf, _Float16* __restrict__ Ch, int ldc,
                    const float* __restrict__ bias,
                    const float* __restrict__ r2a, const float* __restrict__ r2b,
                    const float* __restrict__ Wext, int ldw) {
    constexpr int K = KF * 32;
    const int m0     = blockIdx.x * 16;
    const int wave   = threadIdx.x >> 5;
    const int lane   = threadIdx.x & 31;
    const int laneLo = lane & 15;
    const int laneHi = lane >> 4;            // 0 or 1
    const int kbA    = laneHi * 8;           // A-fragment K base within 16
    const int kbB    = laneHi * 8;           // B-fragment K base within 16
    const int row_m  = m0 + laneLo;          // A row owned by this lane
    const int NT     = Nout >> 4;

    // ---- load all A fragments for this row into registers ----
    AFrag a[KF];
#pragma unroll
    for (int kf = 0; kf < KF; ++kf) {
        const int k0 = kf * 32;
        if constexpr (A16) {
            const _Float16* ap = (const _Float16*)Aptr + (size_t)row_m * lda + k0 + kbA;
            a[kf].q[0] = *(const uint4*)(ap);        // K = k0+kbA .. +7
            a[kf].q[1] = *(const uint4*)(ap + 16);   // K = k0+kbA+16 .. +7
        } else {
            const float* ap = (const float*)Aptr + (size_t)row_m * lda + k0 + kbA;
            float4 f0 = *(const float4*)(ap);
            float4 f1 = *(const float4*)(ap + 4);
            float4 f2 = *(const float4*)(ap + 16);
            float4 f3 = *(const float4*)(ap + 20);
            a[kf].v = (v16h){(_Float16)f0.x, (_Float16)f0.y, (_Float16)f0.z, (_Float16)f0.w,
                             (_Float16)f1.x, (_Float16)f1.y, (_Float16)f1.z, (_Float16)f1.w,
                             (_Float16)f2.x, (_Float16)f2.y, (_Float16)f2.z, (_Float16)f2.w,
                             (_Float16)f3.x, (_Float16)f3.y, (_Float16)f3.z, (_Float16)f3.w};
        }
    }

    for (int nt = wave; nt < NT; nt += 8) {
        const int n0 = nt * 16;
        const _Float16* brow = Bt + (size_t)(n0 + laneLo) * K + kbB;
        // ---- stage all B fragments (deep load clause, staggered waits) ----
        AFrag b[KF];
#pragma unroll
        for (int kf = 0; kf < KF; ++kf) {
            b[kf].q[0] = *(const uint4*)(brow + kf * 32);      // K = k0+kbB .. +7
            b[kf].q[1] = *(const uint4*)(brow + kf * 32 + 16); // K = k0+kbB+16 .. +7
        }
        v8f c = {};
#pragma unroll
        for (int kf = 0; kf < KF; ++kf)
            c = __builtin_amdgcn_wmma_f32_16x16x32_f16(
                    false, a[kf].v, false, b[kf].v, (short)0, c, false, false);

        const int col = n0 + laneLo;
        float bv = 0.0f, w128 = 0.0f, w129 = 0.0f;
        if constexpr (BIAS) bv = bias[col];
        if constexpr (R2) {
            w128 = Wext[(size_t)128 * ldw + col];
            w129 = Wext[(size_t)129 * ldw + col];
        }
        // guard-free epilogue: M is a multiple of 16
#pragma unroll
        for (int r = 0; r < 8; ++r) {
            const int row = m0 + r + 8 * laneHi;
            float v = c[r] + bv;
            if constexpr (R2) v += r2a[row] * w128 + r2b[row] * w129;
            if constexpr (RELU) v = fmaxf(v, 0.0f);
            if constexpr (OUT16) Ch[(size_t)row * ldc + col] = (_Float16)v;
            else                 Cf[(size_t)row * ldc + col] = v;
        }
    }
}

// ---------------------------------------------------------------------------
// GCN aggregation: acc[i] = hlin[i]*dis[i]^2  (self loop), then per-edge
// acc[dst] += hlin[src]*wnorm[e]   (f32 atomics, L2-resident acc)
// W4 = row width in float4s (compile-time so /,% become shifts/mults)
// ---------------------------------------------------------------------------
template<int W4>
__global__ void ENVM_init_acc(const float* __restrict__ hlin, const float* __restrict__ dis,
                              float* __restrict__ acc, int n) {
    long t = (long)blockIdx.x * blockDim.x + threadIdx.x;
    if (t >= (long)n * W4) return;
    int i = (int)(t / W4);
    float s = dis[i]; s = s * s;
    float4 v = ((const float4*)hlin)[t];
    v.x *= s; v.y *= s; v.z *= s; v.w *= s;
    ((float4*)acc)[t] = v;
}

template<int W4>
__global__ void ENVM_edge_agg(const int* __restrict__ src, const int* __restrict__ dst,
                              const float* __restrict__ wnorm, const float* __restrict__ hlin,
                              float* __restrict__ acc, int e) {
    long t = (long)blockIdx.x * blockDim.x + threadIdx.x;
    if (t >= (long)e * W4) return;
    int ed = (int)(t / W4);
    int c  = (int)(t - (long)ed * W4);
    int s = src[ed], d = dst[ed];
    float w = wnorm[ed];
    float4 v = ((const float4*)hlin)[(size_t)s * W4 + c];
    float* o = acc + ((size_t)d * W4 + c) * 4;
    atomicAdd(o + 0, v.x * w);
    atomicAdd(o + 1, v.y * w);
    atomicAdd(o + 2, v.z * w);
    atomicAdd(o + 3, v.w * w);
}

// x3 = (f16) relu(acc + bg1),  width 256
__global__ void ENVM_fin1(const float* __restrict__ acc, const float* __restrict__ bg1,
                          _Float16* __restrict__ x3, int n) {
    long t = (long)blockIdx.x * blockDim.x + threadIdx.x;
    if (t >= (long)n * 256) return;
    int j = (int)(t & 255);
    x3[t] = (_Float16)fmaxf(acc[t] + bg1[j], 0.0f);
}

// hidden = (f16)(mean + noise * exp(clip(logvar,-5,5)))
__global__ void ENVM_hidden(const float* __restrict__ x3h, const float* __restrict__ noise,
                            _Float16* __restrict__ hidden, int n) {
    long t = (long)blockIdx.x * blockDim.x + threadIdx.x;
    if (t >= (long)n * DH) return;
    int i = (int)(t >> 7);
    int j = (int)(t & 127);
    float mean   = x3h[(size_t)i * 256 + j];
    float logvar = x3h[(size_t)i * 256 + 128 + j];
    logvar = fminf(fmaxf(logvar, -5.0f), 5.0f);
    hidden[t] = (_Float16)(mean + noise[t] * __expf(logvar));
}

// g2 = relu(acc2 + bg2); out[:, :128] then out[:, 128] concatenated
__global__ void ENVM_fin2(const float* __restrict__ acc2, const float* __restrict__ bg2,
                          float* __restrict__ out, int n) {
    long t = (long)blockIdx.x * blockDim.x + threadIdx.x;
    if (t >= (long)n * 129) return;
    int i = (int)(t / 129);
    int j = (int)(t - (long)i * 129);
    float v = fmaxf(acc2[(size_t)i * 144 + j] + bg2[j], 0.0f);
    if (j < 128) out[(size_t)i * 128 + j] = v;
    else         out[(size_t)NNODES * 128 + i] = v;
}

// ---------------------------------------------------------------------------
extern "C" void kernel_launch(void* const* d_in, const int* in_sizes, int n_in,
                              void* d_out, int out_size, void* d_ws, size_t ws_size,
                              hipStream_t stream) {
    const float* x      = (const float*)d_in[0];
    const float* rw1    = (const float*)d_in[1];
    const float* action = (const float*)d_in[2];
    const float* noise  = (const float*)d_in[3];
    const int*   eidx   = (const int*)  d_in[4];
    const float* W1     = (const float*)d_in[5];
    const float* b1     = (const float*)d_in[6];
    const float* Wg1    = (const float*)d_in[7];
    const float* bg1    = (const float*)d_in[8];
    const float* Whv    = (const float*)d_in[9];
    const float* bhv    = (const float*)d_in[10];
    const float* Wg2    = (const float*)d_in[11];
    const float* bg2    = (const float*)d_in[12];
    float* out = (float*)d_out;

    const int* src = eidx;
    const int* dst = eidx + NEDGES;

    // ---- workspace layout (256B aligned slabs) ----
    char* ws = (char*)d_ws;
    size_t off = 0;
    auto take = [&](size_t bytes) { char* p = ws + off; off = (off + bytes + 255) & ~(size_t)255; return p; };
    float*     deg   = (float*)    take((size_t)NNODES * 4);
    float*     dis   = (float*)    take((size_t)NNODES * 4);
    float*     wnorm = (float*)    take((size_t)NEDGES * 4);
    _Float16*  W1t   = (_Float16*) take((size_t)128 * 256 * 2);   // [128][256]
    _Float16*  Wg1t  = (_Float16*) take((size_t)256 * 128 * 2);   // [256][128]
    _Float16*  Whvt  = (_Float16*) take((size_t)256 * 256 * 2);   // [256][256]
    _Float16*  Wg2t  = (_Float16*) take((size_t)144 * 128 * 2);   // [144][128] zero-padded
    _Float16*  h     = (_Float16*) take((size_t)NNODES * 128 * 2);  // also 'hidden' later
    float*     hlin  = (float*)    take((size_t)NNODES * 256 * 4);  // also 'x3h' later
    float*     acc1  = (float*)    take((size_t)NNODES * 256 * 4);  // also 'hlin2' (stride 144) later
    _Float16*  x3    = (_Float16*) take((size_t)NNODES * 256 * 2);
    float*     acc2  = (float*)    take((size_t)NNODES * 144 * 4);
    (void)ws_size;

    float*    x3h    = hlin;   // reuse (hlin dead after aggregation 1)
    _Float16* hidden = h;      // reuse (h dead after GEMM2)
    float*    hlin2  = acc1;   // reuse (acc1 dead after fin1)

    const int B = 256;
    auto blocks = [](long total, int b) { return (unsigned)((total + b - 1) / b); };

    // ---- weights: transpose + convert to f16 ----
    ENVM_wt_f16<<<blocks(128L * 256, B), B, 0, stream>>>(W1,  W1t,  256, 128, 128);
    ENVM_wt_f16<<<blocks(256L * 128, B), B, 0, stream>>>(Wg1, Wg1t, 128, 256, 256); // rows 0..127 only
    ENVM_wt_f16<<<blocks(256L * 256, B), B, 0, stream>>>(Whv, Whvt, 256, 256, 256);
    ENVM_wt_f16<<<blocks(144L * 128, B), B, 0, stream>>>(Wg2, Wg2t, 128, 129, 144);

    // ---- degrees / normalization ----
    ENVM_deg_init <<<blocks(NNODES, B), B, 0, stream>>>(deg, NNODES);
    ENVM_deg_edges<<<blocks(NEDGES, B), B, 0, stream>>>(dst, deg, NEDGES);
    ENVM_dis      <<<blocks(NNODES, B), B, 0, stream>>>(deg, dis, NNODES);
    ENVM_enorm    <<<blocks(NEDGES, B), B, 0, stream>>>(src, dst, dis, wnorm, NEDGES);

    const unsigned MT = NNODES / 16;  // 3125 row tiles (exact)

    // ---- GEMM1: h = relu(x @ W1 + b1), f32 A converted on load, f16 out ----
    ENVM_gemm_wmma<false, 8, true, false, true, true><<<MT, B, 0, stream>>>(
        x, 256, W1t, 128, nullptr, h, 128, b1, nullptr, nullptr, nullptr, 0);

    // ---- GEMM2: hlin = [h,rw1,action] @ Wg1  (K=128 WMMA + rank-2 tail) ----
    ENVM_gemm_wmma<true, 4, false, true, false, false><<<MT, B, 0, stream>>>(
        h, 128, Wg1t, 256, hlin, nullptr, 256, nullptr, rw1, action, Wg1, 256);

    // ---- aggregation 1 ----
    ENVM_init_acc<64><<<blocks((long)NNODES * 64, B), B, 0, stream>>>(hlin, dis, acc1, NNODES);
    ENVM_edge_agg<64><<<blocks((long)NEDGES * 64, B), B, 0, stream>>>(src, dst, wnorm, hlin, acc1, NEDGES);
    ENVM_fin1        <<<blocks((long)NNODES * 256, B), B, 0, stream>>>(acc1, bg1, x3, NNODES);

    // ---- GEMM3: x3h = relu(x3 @ Whv + bhv) ----
    ENVM_gemm_wmma<true, 8, true, false, true, false><<<MT, B, 0, stream>>>(
        x3, 256, Whvt, 256, x3h, nullptr, 256, bhv, nullptr, nullptr, nullptr, 0);

    // ---- reparameterize ----
    ENVM_hidden<<<blocks((long)NNODES * 128, B), B, 0, stream>>>(x3h, noise, hidden, NNODES);

    // ---- GEMM4: hlin2 = hidden @ Wg2 (Nout padded 129->144) ----
    ENVM_gemm_wmma<true, 4, false, false, false, false><<<MT, B, 0, stream>>>(
        hidden, 128, Wg2t, 144, hlin2, nullptr, 144, nullptr, nullptr, nullptr, nullptr, 0);

    // ---- aggregation 2 + output ----
    ENVM_init_acc<36><<<blocks((long)NNODES * 36, B), B, 0, stream>>>(hlin2, dis, acc2, NNODES);
    ENVM_edge_agg<36><<<blocks((long)NEDGES * 36, B), B, 0, stream>>>(src, dst, wnorm, hlin2, acc2, NEDGES);
    ENVM_fin2        <<<blocks((long)NNODES * 129, B), B, 0, stream>>>(acc2, bg2, out, NNODES);

    (void)in_sizes; (void)n_in; (void)out_size;
}
